// Discriminator_63660005261324
// MI455X (gfx1250) — matrix operand order repaired and checked by reading, not compiled, not run on hardware
//
#include <hip/hip_runtime.h>

typedef __attribute__((ext_vector_type(16))) _Float16 v16h;
typedef __attribute__((ext_vector_type(2)))  __fp16   v2hh;   // cvt_pkrtz result type
typedef __attribute__((ext_vector_type(8)))  float    v8f;

#define B_TOT   512
#define T_TOT   1024
#define C_CH    16
#define H_DIM   64
#define BT      16        // batch rows per workgroup
#define NTHREADS 256      // 8 wave32
#define TPW     8         // feature tiles (hidden units) per wave

// LDS: AfL 131072 | W0 4096 | bf 4096 | F0/F1/F2 12288 | WinitS 4096
#define SMEM_BYTES 155648

__global__ __launch_bounds__(NTHREADS) void cde_scan_kernel(
    const float* __restrict__ ys,     // [512,1024,16]
    const float* __restrict__ Winit,  // [16,64]
    const float* __restrict__ binit,  // [64]
    const float* __restrict__ Wf,     // [65,1024]
    const float* __restrict__ bfg,    // [1024]
    const float* __restrict__ Wr,     // [64]
    float* __restrict__ partials)     // [32]
{
  extern __shared__ char smem[];
  _Float16* AfL = (_Float16*)smem;                      // W_f^T fragments, swizzled
  float*    W0L = (float*)(smem + 131072);              // W_f row 0 (t-weights)
  float*    bfL = W0L + 1024;
  float*    F0  = bfL + 1024;                           // f triple buffer [16][64]
  float*    F1  = F0  + 1024;
  float*    F2  = F1  + 1024;
  float*    WiS = F2  + 1024;                           // W_init staging [16][64]

  const int tid  = threadIdx.x;
  const int lane = tid & 31;
  const int wv   = tid >> 5;
  const int b0   = blockIdx.x * BT;
  const int m    = lane & 15;         // batch row owned by this lane (B column)
  const int h16  = lane >> 4;         // half-wave
  const int hlo  = 16 * h16;          // lane's h ranges: [hlo,hlo+16) and [hhi,hhi+16)
  const int hhi  = 32 + 16 * h16;

  // ---------- stage weights; build A fragments pre-swizzled into LDS ----------
  // Fragment (global tile tg, kb) for lane L: 16 contiguous f16 at ((tg*2+kb)*32+L)*16
  // element e -> A[row m][K = e+16*h16+32*kb] = Wf[1+K][tg*16+m]
  for (int idx = tid; idx < 64 * 1024; idx += NTHREADS) {
    int e  = idx & 15;
    int L  = (idx >> 4) & 31;
    int g  = idx >> 9;                 // 0..127 = tg*2+kb
    int kb = g & 1;
    int tg = g >> 1;                   // 0..63
    int mm = L & 15, hh = L >> 4;
    int hp = e + 16 * hh + 32 * kb;
    AfL[idx] = (_Float16)Wf[(1 + hp) * 1024 + tg * 16 + mm];
  }
  for (int i = tid; i < 1024; i += NTHREADS) {
    W0L[i] = Wf[i];
    bfL[i] = bfg[i];
    WiS[i] = Winit[i];
  }
  __syncthreads();

  // ---------- h0 into per-lane registers (replicated across waves) ----------
  float Ylo[16], Yhi[16], Hlo[16], Hhi[16];
  {
    const float* yrow = ys + (size_t)(b0 + m) * T_TOT * C_CH;  // Y(t0), row b=m
    #pragma unroll
    for (int j = 0; j < 16; ++j) { Ylo[j] = binit[hlo + j]; Yhi[j] = binit[hhi + j]; }
    #pragma unroll
    for (int c = 0; c < C_CH; ++c) {
      const float yc = yrow[c];
      #pragma unroll
      for (int j = 0; j < 16; ++j) {
        Ylo[j] += yc * WiS[c * H_DIM + hlo + j];
        Yhi[j] += yc * WiS[c * H_DIM + hhi + j];
      }
    }
    #pragma unroll
    for (int j = 0; j < 16; ++j) { Hlo[j] = Ylo[j]; Hhi[j] = Yhi[j]; }
  }

  float* fm2 = F0;   // f_{n-2}
  float* fm1 = F1;   // f_{n-1}
  float* fw  = F2;   // write target f_n

  // ---------- 1024 sequential drift evaluations, ONE barrier per step ----------
  for (int n = 0; n < T_TOT; ++n) {
    const int s = (n < 1022) ? n : 1022;                // clamped segment
    const float tn = (float)n;

    // dY[b=m][c = 8*h16 + q], q=0..7 : direct global, L2-resident
    const float* pr = ys + ((size_t)(b0 + m) * T_TOT + s) * C_CH + 8 * h16;
    float dv[8];
    #pragma unroll
    for (int q = 0; q < 8; ++q) dv[q] = pr[C_CH + q] - pr[q];
    __builtin_prefetch(pr + 2 * C_CH, 0, 0);

    if (n > 0) {  // fused: y += 0.5*(f_{n-2}+f_{n-1}) (n>=2) ; yhat = 2y - yhat + f_{n-1}
      float f1lo[16], f1hi[16];
      const float* p1 = fm1 + m * H_DIM;
      #pragma unroll
      for (int j = 0; j < 16; ++j) { f1lo[j] = p1[hlo + j]; f1hi[j] = p1[hhi + j]; }
      if (n > 1) {
        const float* p2 = fm2 + m * H_DIM;
        #pragma unroll
        for (int j = 0; j < 16; ++j) {
          Ylo[j] += 0.5f * (p2[hlo + j] + f1lo[j]);
          Yhi[j] += 0.5f * (p2[hhi + j] + f1hi[j]);
        }
      }
      #pragma unroll
      for (int j = 0; j < 16; ++j) {
        Hlo[j] = 2.0f * Ylo[j] - Hlo[j] + f1lo[j];
        Hhi[j] = 2.0f * Yhi[j] - Hhi[j] + f1hi[j];
      }
    }

    // B = yhat^T fragments from registers: blo[e] <-> K = hlo+e, bhi[e] <-> hhi+e
    v16h blo, bhi;
    {
      union { v16h v; v2hh p[8]; } u;
      #pragma unroll
      for (int jj = 0; jj < 8; ++jj)
        u.p[jj] = __builtin_amdgcn_cvt_pkrtz(Hlo[2 * jj], Hlo[2 * jj + 1]);
      blo = u.v;
      #pragma unroll
      for (int jj = 0; jj < 8; ++jj)
        u.p[jj] = __builtin_amdgcn_cvt_pkrtz(Hhi[2 * jj], Hhi[2 * jj + 1]);
      bhi = u.v;
    }

    #pragma unroll
    for (int t = 0; t < TPW; ++t) {
      const int gt = wv * TPW + t;
      const int cb = gt * 16 + 8 * h16;                 // feature base for acc rows
      // A fragments streamed from pre-swizzled LDS (two b128 each, conflict-free)
      const v16h a0 = *(const v16h*)(AfL + ((gt * 2 + 0) * 32 + lane) * 16);
      const v16h a1 = *(const v16h*)(AfL + ((gt * 2 + 1) * 32 + lane) * 16);
      v8f acc;
      #pragma unroll
      for (int r = 0; r < 8; ++r) acc[r] = bfL[cb + r] + tn * W0L[cb + r];
      acc = __builtin_amdgcn_wmma_f32_16x16x32_f16(false, a0, false, blo,
                                                   (short)0, acc, false, false);
      acc = __builtin_amdgcn_wmma_f32_16x16x32_f16(false, a1, false, bhi,
                                                   (short)0, acc, false, false);
      // f[b=m, h] = sum_c D[c,b]*dY[b,c] ; rows are c = r + 8*h16 -> in-lane
      float sv = acc[0] * dv[0] + acc[1] * dv[1] + acc[2] * dv[2] + acc[3] * dv[3]
               + acc[4] * dv[4] + acc[5] * dv[5] + acc[6] * dv[6] + acc[7] * dv[7];
      sv += __shfl_xor(sv, 16, 32);          // combine c 0..7 with c 8..15
      fw[m * H_DIM + gt] = sv;               // both halves: same addr, same value
    }
    __syncthreads();                         // publish f_n ; protect next write target
    float* tt = fm2; fm2 = fm1; fm1 = fw; fw = tt;
  }

  // ---------- final y update + readout from wave 0's register copy ----------
  if (wv == 0) {
    const float* p1 = fm1 + m * H_DIM;       // f_1023
    const float* p2 = fm2 + m * H_DIM;       // f_1022
    float pacc = 0.f;
    #pragma unroll
    for (int j = 0; j < 16; ++j) {
      float ylo = Ylo[j] + 0.5f * (p2[hlo + j] + p1[hlo + j]);
      float yhi = Yhi[j] + 0.5f * (p2[hhi + j] + p1[hhi + j]);
      pacc += ylo * Wr[hlo + j] + yhi * Wr[hhi + j];
    }
    pacc += __shfl_xor(pacc, 1, 32);
    pacc += __shfl_xor(pacc, 2, 32);
    pacc += __shfl_xor(pacc, 4, 32);
    pacc += __shfl_xor(pacc, 8, 32);
    pacc += __shfl_xor(pacc, 16, 32);
    if (lane == 0) partials[blockIdx.x] = pacc;
  }
}

__global__ __launch_bounds__(32) void cde_reduce_kernel(
    const float* __restrict__ partials, const float* __restrict__ br,
    float* __restrict__ out)
{
  float v = partials[threadIdx.x];
  v += __shfl_xor(v, 1, 32);
  v += __shfl_xor(v, 2, 32);
  v += __shfl_xor(v, 4, 32);
  v += __shfl_xor(v, 8, 32);
  v += __shfl_xor(v, 16, 32);
  if (threadIdx.x == 0) out[0] = v * (1.0f / (float)B_TOT) + br[0];
}

extern "C" void kernel_launch(void* const* d_in, const int* in_sizes, int n_in,
                              void* d_out, int out_size, void* d_ws, size_t ws_size,
                              hipStream_t stream) {
  (void)in_sizes; (void)n_in; (void)out_size; (void)ws_size;
  const float* ys    = (const float*)d_in[0];
  const float* Winit = (const float*)d_in[1];
  const float* binit = (const float*)d_in[2];
  const float* Wf    = (const float*)d_in[3];
  const float* bfg   = (const float*)d_in[4];
  const float* Wr    = (const float*)d_in[5];
  const float* br    = (const float*)d_in[6];
  float* partials = (float*)d_ws;

  (void)hipFuncSetAttribute((const void*)cde_scan_kernel,
                            hipFuncAttributeMaxDynamicSharedMemorySize,
                            SMEM_BYTES);

  cde_scan_kernel<<<B_TOT / BT, NTHREADS, SMEM_BYTES, stream>>>(
      ys, Winit, binit, Wf, bfg, Wr, partials);
  cde_reduce_kernel<<<1, 32, 0, stream>>>(partials, br, (float*)d_out);
}
